// Router_45028437131468
// MI455X (gfx1250) — matrix operand order repaired and checked by reading, not compiled, not run on hardware
//
#include <hip/hip_runtime.h>
#include <hip/hip_bf16.h>

// ---------------------------------------------------------------------------
// MoE router for MI455X (gfx1250, wave32, WMMA).
//   x[8,4096,1024] f32, W[1024,64] f32, bias[64] f32, temperature[1] f32,
//   expert_capacity int (1024).
// Outputs (concatenated as f32): indices[64,cap,2], scores[64,cap], loss[1].
// ---------------------------------------------------------------------------

typedef __attribute__((ext_vector_type(16))) __bf16 bf16x16;
typedef __attribute__((ext_vector_type(2)))  __bf16 bf16x2;
typedef __attribute__((ext_vector_type(8)))  float  v8f;
typedef int v4i_vs __attribute__((vector_size(16)));   // int4 vector for async builtin

#define TOKENS      32768
#define DMODEL      1024
#define NEXPERT     64
#define GROUP_SIZE  4096
#define KSTEPS      32          // 1024 / 32
#define KS_PER_PHASE 8
#define NPHASE      (KSTEPS / KS_PER_PHASE)     // 4
#define TILES       (TOKENS / 16)
#define WAVES_PER_BLOCK 8
#define GEMM_BLOCKS (TILES / WAVES_PER_BLOCK)   // 256
#define NPART       67          // 64 usage + ent + conf + z

union ABu { bf16x16 v; unsigned u[8]; };

// f32 pair -> packed bf16 (RNE). Use native __bf16 fptrunc so the backend can
// select the hardware convert (v_cvt_pk_bf16_f32 when available) instead of a
// hand-rolled integer rounding sequence.
__device__ __forceinline__ unsigned pack_bf16_rne(float a, float b) {
  bf16x2 t = { (__bf16)a, (__bf16)b };
  return __builtin_bit_cast(unsigned, t);
}

// 16-byte global -> LDS copy; async (ASYNCcnt-tracked) when available.
__device__ __forceinline__ void copy16_g2l(const unsigned* gsrc, unsigned* ldst) {
#if __has_builtin(__builtin_amdgcn_global_load_async_to_lds_b128)
  __builtin_amdgcn_global_load_async_to_lds_b128(
      (__attribute__((address_space(1))) v4i_vs*)const_cast<unsigned*>(gsrc),
      (__attribute__((address_space(3))) v4i_vs*)ldst, 0, 0);
#else
  *(uint4*)ldst = *(const uint4*)gsrc;
#endif
}

__device__ __forceinline__ void copy_g2l_fence() {
#if __has_builtin(__builtin_amdgcn_global_load_async_to_lds_b128)
  // ASYNCcnt is NOT covered by __syncthreads(); drain our async copies first.
  asm volatile("s_wait_asynccnt 0x0" ::: "memory");
#endif
}

// ---------------------------------------------------------------------------
// K0: repack W[1024,64] f32 into bf16 B fragments matching the CDNA5
// 16-bit B-matrix (32x16) VGPR layout:
//   VGPR v, lanes 0-15 : K = 2v, 2v+1 (v<4); 8+2v, 8+2v+1 (v>=4)   (column N = lane)
//   VGPR v, lanes 16-31: K = 8+2v, 8+2v+1 (v<4); 16+2v, 16+2v+1 (v>=4)
// Bp layout: [et(4)][ks(32)][lane(32)][v(8)] uints.
// ---------------------------------------------------------------------------
__global__ __launch_bounds__(256) void pack_kernel(const float* __restrict__ W,
                                                   unsigned* __restrict__ Bp) {
  int id = blockIdx.x * 256 + threadIdx.x;          // 0 .. 32767
  int v    = id & 7;
  int lane = (id >> 3) & 31;
  int ks   = (id >> 8) & 31;
  int et   = id >> 13;
  int n    = lane & 15;
  int koff = ((lane < 16) ? 0 : 8) + ((v >= 4) ? 8 : 0);
  int k    = ks * 32 + koff + 2 * v;
  float f0 = W[(size_t)k * NEXPERT + et * 16 + n];
  float f1 = W[(size_t)(k + 1) * NEXPERT + et * 16 + n];
  Bp[id] = pack_bf16_rne(f0, f1);
}

// ---------------------------------------------------------------------------
// K1: GEMM (bf16 WMMA, f32 accum) + bias + temperature softmax + loss partials
//     + transposed prob store probsT[E][TOKENS].
// One wave per 16-token tile; 8 waves / block. B fragments are staged to LDS
// once per block (4 x 32KB phases, async-to-LDS when available) so the VMEM
// pipe carries only the HBM stream of x; B comes from conflict-free ds loads.
// ---------------------------------------------------------------------------
__global__ __launch_bounds__(256) void gemm_softmax_kernel(
    const float* __restrict__ x, const unsigned* __restrict__ Bp,
    const float* __restrict__ bias, const float* __restrict__ temp,
    float* __restrict__ probsT, float* __restrict__ partials) {

  __shared__ unsigned Bs[4 * KS_PER_PHASE * 32 * 8];   // 8192 uints = 32 KB
  __shared__ float wUsage[WAVES_PER_BLOCK][NEXPERT];
  __shared__ float wScal[WAVES_PER_BLOCK][3];

  const int tid  = threadIdx.x;
  const int wave = tid >> 5;
  const int lane = tid & 31;
  const int m16  = lane & 15;
  const int half = lane >> 4;

  const int tile     = blockIdx.x * WAVES_PER_BLOCK + wave;
  const int tileBase = tile * 16;

  // A row for this lane: M = lane & 15 (both lane halves cover same 16 rows).
  const float* xrow = x + (size_t)(tileBase + m16) * DMODEL;

  v8f acc[4];
  #pragma unroll
  for (int et = 0; et < 4; ++et) acc[et] = (v8f){0.f,0.f,0.f,0.f,0.f,0.f,0.f,0.f};

  for (int phase = 0; phase < NPHASE; ++phase) {
    // ---- stage this phase's B fragments (all 4 expert tiles, 8 K-steps) ----
    #pragma unroll
    for (int et = 0; et < 4; ++et) {
      const unsigned* gsrc = Bp + (size_t)(et * KSTEPS + phase * KS_PER_PHASE) * 256;
      unsigned* ldst = Bs + et * (KS_PER_PHASE * 256);
      #pragma unroll
      for (int c = 0; c < 2; ++c) {
        int q = tid + c * 256;                     // uint4 index, 512 per chunk
        copy16_g2l(gsrc + q * 4, ldst + q * 4);
      }
    }
    copy_g2l_fence();
    __syncthreads();

    for (int ksl = 0; ksl < KS_PER_PHASE; ++ksl) {
      const int ks = phase * KS_PER_PHASE + ksl;
      // --- A fragment: 16-bit A 16x32 layout.
      //   lanes 0-15 : K = {0..7, 16..23} + ks*32 ; lanes 16-31: +8 shift.
      const float* seg0 = xrow + ks * 32 + half * 8;
      float4 a0 = *(const float4*)(seg0);
      float4 a1 = *(const float4*)(seg0 + 4);
      float4 a2 = *(const float4*)(seg0 + 16);
      float4 a3 = *(const float4*)(seg0 + 20);
      __builtin_prefetch((const void*)(seg0 + 32), 0, 1);   // next K-step of A

      ABu a;
      a.u[0] = pack_bf16_rne(a0.x, a0.y); a.u[1] = pack_bf16_rne(a0.z, a0.w);
      a.u[2] = pack_bf16_rne(a1.x, a1.y); a.u[3] = pack_bf16_rne(a1.z, a1.w);
      a.u[4] = pack_bf16_rne(a2.x, a2.y); a.u[5] = pack_bf16_rne(a2.z, a2.w);
      a.u[6] = pack_bf16_rne(a3.x, a3.y); a.u[7] = pack_bf16_rne(a3.z, a3.w);

      #pragma unroll
      for (int et = 0; et < 4; ++et) {
        const unsigned* bp = Bs + et * (KS_PER_PHASE * 256) + (ksl * 32 + lane) * 8;
        ABu b;
        *(uint4*)&b.u[0] = *(const uint4*)(bp);
        *(uint4*)&b.u[4] = *(const uint4*)(bp + 4);
        acc[et] = __builtin_amdgcn_wmma_f32_16x16x32_bf16(
            false, a.v, false, b.v, (short)0, acc[et], false, false);
      }
    }
    __syncthreads();   // Bs reused next phase
  }

  // --- Epilogue: bias + softmax(l / safeT) + losses, per token row.
  // C layout: VGPR r, lanes 0-15 -> (M=r, N=lane); lanes 16-31 -> (M=r+8).
  float bias4[4];
  #pragma unroll
  for (int et = 0; et < 4; ++et) bias4[et] = bias[et * 16 + m16];

  const float safeT = fmaxf(temp[0], 0.1f);
  const float invT  = 1.0f / safeT;

  float usagePart[4] = {0.f, 0.f, 0.f, 0.f};
  float entAcc = 0.f, confAcc = 0.f, zAcc = 0.f;

  #pragma unroll
  for (int r = 0; r < 8; ++r) {
    float l[4];
    #pragma unroll
    for (int et = 0; et < 4; ++et) l[et] = acc[et][r] + bias4[et];

    float mx = fmaxf(fmaxf(l[0], l[1]), fmaxf(l[2], l[3]));
    #pragma unroll
    for (int m = 1; m <= 8; m <<= 1) mx = fmaxf(mx, __shfl_xor(mx, m));

    float e[4];
    float st = 0.f, s1 = 0.f;
    #pragma unroll
    for (int et = 0; et < 4; ++et) {
      e[et] = __expf((l[et] - mx) * invT);
      st += e[et];
      s1 += __expf(l[et] - mx);          // raw logsumexp for z-loss
    }
    #pragma unroll
    for (int m = 1; m <= 8; m <<= 1) { st += __shfl_xor(st, m); s1 += __shfl_xor(s1, m); }

    const float rS = 1.0f / st;          // == max prob (exp(0)/st)
    const int token = tileBase + r + 8 * half;

    float ep = 0.f;
    #pragma unroll
    for (int et = 0; et < 4; ++et) {
      float p = e[et] * rS;
      ep += p * __logf(p + 1e-5f);
      usagePart[et] += p;
      probsT[(size_t)(et * 16 + m16) * TOKENS + token] = p;
    }
    #pragma unroll
    for (int m = 1; m <= 8; m <<= 1) ep += __shfl_xor(ep, m);

    entAcc  += -ep;
    confAcc += 1.0f - rS;
    const float lse = mx + __logf(s1);
    zAcc    += lse * lse;
  }

  // combine half-waves (lane L and L+16 hold disjoint token sets)
  #pragma unroll
  for (int et = 0; et < 4; ++et) usagePart[et] += __shfl_xor(usagePart[et], 16);
  entAcc  += __shfl_xor(entAcc, 16);
  confAcc += __shfl_xor(confAcc, 16);
  zAcc    += __shfl_xor(zAcc, 16);

  if (half == 0) {
    #pragma unroll
    for (int et = 0; et < 4; ++et) wUsage[wave][et * 16 + m16] = usagePart[et];
  }
  if (lane == 0) { wScal[wave][0] = entAcc; wScal[wave][1] = confAcc; wScal[wave][2] = zAcc; }
  __syncthreads();

  // deterministic per-block partials (fixed summation order; no float atomics)
  if (tid < 64) {
    float s = 0.f;
    #pragma unroll
    for (int w = 0; w < WAVES_PER_BLOCK; ++w) s += wUsage[w][tid];
    partials[blockIdx.x * NPART + tid] = s;
  } else if (tid < NPART) {
    float s = 0.f;
    #pragma unroll
    for (int w = 0; w < WAVES_PER_BLOCK; ++w) s += wScal[w][tid - 64];
    partials[blockIdx.x * NPART + tid] = s;
  }
}

// ---------------------------------------------------------------------------
// K3: expert-choice top-k. One block (1024 threads) per expert row.
// Each thread keeps its 32 row values in registers (single coalesced sweep),
// then: 4-pass radix select on positive-float bit patterns (order preserving)
// -> exact k-th threshold; gather >= T; bitonic sort (score desc, index asc),
// matching jax.lax.top_k tie-breaking; emit (group,pos) + scores.
// ---------------------------------------------------------------------------
#define TK_CAND 2048

__global__ __launch_bounds__(1024) void topk_kernel(
    const float* __restrict__ probsT, const int* __restrict__ capPtr,
    float* __restrict__ out) {

  __shared__ unsigned hist[256];
  __shared__ unsigned s_prefix;
  __shared__ int      s_k;
  __shared__ unsigned s_cnt;
  __shared__ float    ss[TK_CAND];
  __shared__ int      sidx[TK_CAND];

  const int e   = blockIdx.x;
  const int tid = threadIdx.x;
  const float* row = probsT + (size_t)e * TOKENS;

  int cap = *capPtr;
  if (cap > TOKENS) cap = TOKENS;
  if (cap > 1024)   cap = 1024;

  // ---- single global sweep: 32 values per thread into registers ----
  // value t = vals[p*4 + j] corresponds to row index tid*4 + p*4096 + j.
  float vals[32];
  #pragma unroll
  for (int p = 0; p < 8; ++p) {
    float4 v4 = *(const float4*)(row + p * 4096 + tid * 4);
    vals[p * 4 + 0] = v4.x; vals[p * 4 + 1] = v4.y;
    vals[p * 4 + 2] = v4.z; vals[p * 4 + 3] = v4.w;
  }

  if (tid == 0) { s_prefix = 0u; s_k = cap; }
  __syncthreads();

  // ---- 4-pass radix select (8 bits / pass, MSB first), from registers ----
  for (int pass = 0; pass < 4; ++pass) {
    const int shift = 24 - pass * 8;
    if (tid < 256) hist[tid] = 0u;
    __syncthreads();
    const unsigned pfx   = s_prefix;
    const unsigned pmask = (pass == 0) ? 0u : (0xFFFFFFFFu << (shift + 8));
    #pragma unroll
    for (int t = 0; t < 32; ++t) {
      unsigned key = __float_as_uint(vals[t]);
      if ((key & pmask) == pfx) atomicAdd(&hist[(key >> shift) & 255u], 1u);
    }
    __syncthreads();
    if (tid == 0) {
      int k = s_k;
      unsigned acc = 0;
      int b = 255;
      for (; b > 0; --b) {
        if (acc + hist[b] >= (unsigned)k) break;
        acc += hist[b];
      }
      s_k = k - (int)acc;
      s_prefix = s_prefix | ((unsigned)b << shift);
    }
    __syncthreads();
  }
  const unsigned T = s_prefix;   // exact bit pattern of the cap-th value

  // ---- gather all candidates >= T ----
  if (tid == 0) s_cnt = 0u;
  __syncthreads();
  #pragma unroll
  for (int p = 0; p < 8; ++p) {
    #pragma unroll
    for (int j = 0; j < 4; ++j) {
      float v = vals[p * 4 + j];
      unsigned key = __float_as_uint(v);
      if (key >= T) {
        unsigned slot = atomicAdd(&s_cnt, 1u);
        if (slot < (unsigned)TK_CAND) { ss[slot] = v; sidx[slot] = tid * 4 + p * 4096 + j; }
      }
    }
  }
  __syncthreads();
  unsigned n = s_cnt; if (n > TK_CAND) n = TK_CAND;
  for (int i = tid; i < TK_CAND; i += 1024)
    if (i >= (int)n) { ss[i] = -1.0f; sidx[i] = 0x7FFFFFFF; }
  __syncthreads();

  // ---- bitonic sort, ordering: score desc, tie -> index asc ----
  for (int k = 2; k <= TK_CAND; k <<= 1) {
    for (int j = k >> 1; j > 0; j >>= 1) {
      for (int i = tid; i < TK_CAND; i += 1024) {
        int ixj = i ^ j;
        if (ixj > i) {
          bool up = ((i & k) == 0);
          float sa = ss[i], sb = ss[ixj];
          int   ia = sidx[i], ib = sidx[ixj];
          bool precBA = (sb > sa) || (sb == sa && ib < ia);  // ixj should come first
          if (up ? precBA : !precBA) {
            ss[i] = sb; ss[ixj] = sa; sidx[i] = ib; sidx[ixj] = ia;
          }
        }
      }
      __syncthreads();
    }
  }

  // ---- emit ----
  float* outIdx   = out;                                   // [E, cap, 2]
  float* outScore = out + (size_t)NEXPERT * cap * 2;       // [E, cap]
  for (int t = tid; t < cap; t += 1024) {
    int token = sidx[t];
    float sc  = ss[t];
    if (token == 0x7FFFFFFF) { token = 0; sc = 0.f; }      // safety (shouldn't hit)
    int g = token >> 12;           // / 4096
    int p = token & 4095;          // % 4096
    outIdx[((size_t)e * cap + t) * 2 + 0] = (float)g;
    outIdx[((size_t)e * cap + t) * 2 + 1] = (float)p;
    outScore[(size_t)e * cap + t] = sc;
  }
}

// ---------------------------------------------------------------------------
// K2: deterministic final loss reduction.
// ---------------------------------------------------------------------------
__global__ __launch_bounds__(64) void loss_kernel(const float* __restrict__ partials,
                                                  const int* __restrict__ capPtr,
                                                  float* __restrict__ out) {
  __shared__ float usage[NEXPERT];
  const int tid = threadIdx.x;
  float u = 0.f;
  for (int b = 0; b < GEMM_BLOCKS; ++b) u += partials[b * NPART + tid];
  usage[tid] = u;
  __syncthreads();
  if (tid == 0) {
    float ent = 0.f, conf = 0.f, z = 0.f;
    for (int b = 0; b < GEMM_BLOCKS; ++b) {
      ent  += partials[b * NPART + 64];
      conf += partials[b * NPART + 65];
      z    += partials[b * NPART + 66];
    }
    const float total = (float)TOKENS;
    float usage_loss = 0.f;
    for (int ee = 0; ee < NEXPERT; ++ee) {
      float ue = usage[ee] / (total + 1e-5f);
      float d  = ue - 1.0f / NEXPERT;
      usage_loss += d * d;
    }
    usage_loss /= (float)NEXPERT;
    float mean_ent  = ent / total;
    float ent_loss  = fabsf(mean_ent - 0.5f * __logf((float)NEXPERT));
    float conf_loss = conf / total;
    float z_loss    = z / total;
    int cap = *capPtr; if (cap > TOKENS) cap = TOKENS; if (cap > 1024) cap = 1024;
    out[(size_t)NEXPERT * cap * 3] =
        usage_loss * 0.01f + conf_loss * 0.0001f + ent_loss * 0.0001f + z_loss * 0.001f;
  }
}

// ---------------------------------------------------------------------------
extern "C" void kernel_launch(void* const* d_in, const int* in_sizes, int n_in,
                              void* d_out, int out_size, void* d_ws, size_t ws_size,
                              hipStream_t stream) {
  const float* x    = (const float*)d_in[0];
  const float* W    = (const float*)d_in[1];
  const float* bias = (const float*)d_in[2];
  const float* temp = (const float*)d_in[3];
  const int*   cap  = (const int*)d_in[4];
  float* out = (float*)d_out;

  char* ws = (char*)d_ws;
  float*    probsT   = (float*)ws;                         // 8 MB
  unsigned* Bp       = (unsigned*)(ws + (size_t)8388608);  // 128 KB
  float*    partials = (float*)(ws + (size_t)8519680);     // 256*67*4

  pack_kernel<<<128, 256, 0, stream>>>(W, Bp);
  gemm_softmax_kernel<<<GEMM_BLOCKS, 256, 0, stream>>>(x, Bp, bias, temp, probsT, partials);
  topk_kernel<<<NEXPERT, 1024, 0, stream>>>(probsT, cap, out);
  loss_kernel<<<1, 64, 0, stream>>>(partials, cap, out);
}
